// HyperGNNBinaryGraph_15951508537730
// MI455X (gfx1250) — matrix operand order repaired and checked by reading, not compiled, not run on hardware
//
#include <hip/hip_runtime.h>
#include <hip/hip_bf16.h>

// Problem constants (from reference): BS=4, NE=40, DM=DR=128
#define BSZ  4
#define NE   40
#define DMD  128
#define DRD  128
#define KTOT (2 * NE)   // 80 attention slots
#define LSTR 136        // padded LDS stride in halves (272B = 17*16B -> conflict-free frags, 16B aligned)

typedef _Float16 v8h  __attribute__((ext_vector_type(8)));
typedef _Float16 v16h __attribute__((ext_vector_type(16)));
typedef float    v8f  __attribute__((ext_vector_type(8)));

static __device__ __forceinline__ float fast_exp2(float x) {
#if __has_builtin(__builtin_amdgcn_exp2f)
    return __builtin_amdgcn_exp2f(x);      // v_exp_f32
#else
    return exp2f(x);
#endif
}
static __device__ __forceinline__ float fast_rcp(float x) {
#if __has_builtin(__builtin_amdgcn_rcpf)
    return __builtin_amdgcn_rcpf(x);       // v_rcp_f32
#else
    return 1.0f / x;
#endif
}

// Branchless tanh-form GELU: x * sigmoid(2*0.7978845608*(x + 0.044715 x^3)).
// exp2 constant folds in log2(e):  2*0.7978845608*log2(e) = 2.3022082571
static __device__ __forceinline__ float gelu_fast(float x) {
    float t = x * x;
    float s = __builtin_fmaf(t, 0.044715f, 1.0f);   // 1 + 0.044715 x^2
    float z = (x * s) * -2.3022082571f;             // -2y * log2(e)
    return x * fast_rcp(1.0f + fast_exp2(z));       // x -> x (z->-inf), 0 (z->+inf)
}

// Load a 16-half WMMA fragment slice: halves [kb..kb+7] and [kb+16..kb+23] of one LDS row.
// Matches CDNA5 16-bit A/B layout: lanes 0-15 carry K=0..7,16..23; lanes 16-31 carry K=8..15,24..31
// (caller folds the +8 for the upper lane half into kb).
static __device__ __forceinline__ v16h ld_frag16(const _Float16* p, int kb) {
    v8h lo = *(const v8h*)(p + kb);
    v8h hi = *(const v8h*)(p + kb + 16);
    return __builtin_shufflevector(lo, hi, 0, 1, 2, 3, 4, 5, 6, 7,
                                           8, 9, 10, 11, 12, 13, 14, 15);
}

__global__ __launch_bounds__(256)
void sib_attn_fused(const float* __restrict__ ht,
                    const float* __restrict__ factor,
                    const int*   __restrict__ entn,
                    const float* __restrict__ Wcomb,
                    const float* __restrict__ bcomb,
                    const float* __restrict__ wv,
                    const float* __restrict__ Wfc,
                    const float* __restrict__ bfc,
                    const float* __restrict__ lns,
                    const float* __restrict__ lnb,
                    float* __restrict__ out)
{
    __shared__ __align__(16) _Float16 A16[KTOT * LSTR];   // total_h tile, f16 (WMMA A operand)
    __shared__ __align__(16) _Float16 Bt [DMD  * LSTR];   // Wc_f transposed: Bt[e][k] (WMMA B operand)
    __shared__ float A32[KTOT * DMD];                     // total_h tile, f32 (attention output path)
    __shared__ float ht_s[DMD], hp_s[DMD], wv_s[DMD], o_s[DMD];
    __shared__ float energy_s[KTOT];
    __shared__ float attn_s[KTOT];
    __shared__ float red[256];
    __shared__ float stat[2];

    const int tid = threadIdx.x;
    const int bij = blockIdx.x;                 // 0 .. 6399
    const int b   = bij / (NE * NE);
    const int ij  = bij - b * NE * NE;
    const int i   = ij / NE;
    const int j   = ij - i * NE;

    const size_t rowOff = (size_t)bij * DMD;                          // ht / out row
    const size_t fBase  = ((size_t)(b * NE + i)) * NE * NE * DMD;     // factor[b,i,0,0,0]
    const float* f_j    = factor + fBase + (size_t)j * NE * DMD;      // factor[b,i,j,:,:] (contiguous)

    // ---------------- Phase 0: stage tiles into LDS -------------------------
    // Rows 0..39 of total_h: contiguous factor[b,i,j,k,:]
    {
        const float4* f4 = (const float4*)f_j;
        for (int idx = tid; idx < NE * DMD / 4; idx += 256) {
            float4 v = f4[idx];
            int k = idx >> 5;               // 32 float4 per 128-wide row
            int d = (idx & 31) * 4;
            float* a32 = &A32[k * DMD + d];
            a32[0] = v.x; a32[1] = v.y; a32[2] = v.z; a32[3] = v.w;
            _Float16* a16 = &A16[k * LSTR + d];
            a16[0] = (_Float16)v.x; a16[1] = (_Float16)v.y;
            a16[2] = (_Float16)v.z; a16[3] = (_Float16)v.w;
        }
    }
    // Rows 40..79 of total_h: transposed factor[b,i,k,j,:]
    for (int idx = tid; idx < NE * DMD / 4; idx += 256) {
        int k = idx >> 5;
        int d = (idx & 31) * 4;
        float4 v = *(const float4*)(factor + fBase + (size_t)k * NE * DMD + (size_t)j * DMD + d);
        int kk = NE + k;
        float* a32 = &A32[kk * DMD + d];
        a32[0] = v.x; a32[1] = v.y; a32[2] = v.z; a32[3] = v.w;
        _Float16* a16 = &A16[kk * LSTR + d];
        a16[0] = (_Float16)v.x; a16[1] = (_Float16)v.y;
        a16[2] = (_Float16)v.z; a16[3] = (_Float16)v.w;
    }
    // Wc_f = W_comb[128:,:]  -> transposed into Bt[e][k]
    {
        const float4* w4 = (const float4*)(Wcomb + (size_t)DRD * DMD);
        for (int idx = tid; idx < DMD * DMD / 4; idx += 256) {
            float4 v = w4[idx];
            int k = idx >> 5;
            int e = (idx & 31) * 4;
            Bt[(e + 0) * LSTR + k] = (_Float16)v.x;
            Bt[(e + 1) * LSTR + k] = (_Float16)v.y;
            Bt[(e + 2) * LSTR + k] = (_Float16)v.z;
            Bt[(e + 3) * LSTR + k] = (_Float16)v.w;
        }
    }
    if (tid < DMD) { ht_s[tid] = ht[rowOff + tid]; wv_s[tid] = wv[tid]; }
    if (tid < KTOT) energy_s[tid] = 0.0f;
    __syncthreads();

    // ---------------- Phase 1: ht_proj row (1x128 @ 128x128) ----------------
    if (tid < DMD) {
        float hp = bcomb[tid];
        for (int d = 0; d < DMD; ++d) hp += ht_s[d] * Wcomb[d * DMD + tid];
        hp_s[tid] = hp;
    }
    __syncthreads();

    // ---------------- Phase 2: WMMA GEMM + GELU + energy --------------------
    // pre(80x128) = A16(80x128) @ Wc_f(128x128) + hp ; h = gelu(pre); energy[m] = sum_n h*wv[n]
    {
        const int lane  = tid & 31;
        const int nt    = tid >> 5;               // each of 8 waves owns one 16-col tile
        const int nloc  = lane & 15;
        const int col   = nt * 16 + nloc;
        const int kbsel = (lane >> 4) << 3;       // 0 for lanes 0-15, 8 for lanes 16-31

        v16h bf[4];
        #pragma unroll
        for (int kt = 0; kt < 4; ++kt)
            bf[kt] = ld_frag16(&Bt[col * LSTR], kt * 32 + kbsel);

        const float bias = hp_s[col];
        const float wvc  = wv_s[col];
        const int rowbase0 = (lane >> 4) << 3;    // rows r (lanes 0-15) / 8+r (lanes 16-31)

        #pragma unroll
        for (int mt = 0; mt < 5; ++mt) {
            const _Float16* arow = &A16[(mt * 16 + nloc) * LSTR] + kbsel;
            v8f acc = {0.f, 0.f, 0.f, 0.f, 0.f, 0.f, 0.f, 0.f};
            #pragma unroll
            for (int kt = 0; kt < 4; ++kt) {
                v16h af = ld_frag16(arow, kt * 32);
                acc = __builtin_amdgcn_wmma_f32_16x16x32_f16(
                          false, af, false, bf[kt], (short)0, acc, false, false);
            }
            // in-register epilogue: bias + branchless GELU + scale by w_v[col]
            float p[8];
            #pragma unroll
            for (int r = 0; r < 8; ++r)
                p[r] = gelu_fast(acc[r] + bias) * wvc;
            // reduce across the 16 lanes sharing a row (D layout: VGPR r -> M=r / M=8+r)
            #pragma unroll
            for (int r = 0; r < 8; ++r) {
                #pragma unroll
                for (int m = 1; m < 16; m <<= 1) p[r] += __shfl_xor(p[r], m, 32);
            }
            if (nloc == 0) {
                #pragma unroll
                for (int r = 0; r < 8; ++r)
                    atomicAdd(&energy_s[mt * 16 + rowbase0 + r], p[r]);
            }
        }
    }
    __syncthreads();

    // ---------------- Phase 3: mask + softmax over 80 slots -----------------
    const int nb = entn[b];
    float ev = -3.0e38f;
    if (tid < KTOT) {
        int kk = (tid < NE) ? tid : tid - NE;
        bool validall = (i < nb) && (j < nb) && (kk < nb);
        bool masked = (j == kk) || !validall;
        ev = masked ? -10000.0f : energy_s[tid];
    }
    red[tid] = ev;
    __syncthreads();
    for (int s = 128; s > 0; s >>= 1) {
        if (tid < s) red[tid] = fmaxf(red[tid], red[tid + s]);
        __syncthreads();
    }
    if (tid == 0) stat[0] = red[0];
    __syncthreads();
    float ex = (tid < KTOT) ? __expf(ev - stat[0]) : 0.0f;
    red[tid] = ex;
    __syncthreads();
    for (int s = 128; s > 0; s >>= 1) {
        if (tid < s) red[tid] += red[tid + s];
        __syncthreads();
    }
    if (tid == 0) stat[1] = red[0];
    __syncthreads();
    if (tid < KTOT) attn_s[tid] = ex / stat[1];
    __syncthreads();

    // ---------------- Phase 4: attention-weighted sum (f32 path) ------------
    if (tid < DMD) {
        float o = 0.0f;
        for (int k = 0; k < KTOT; ++k) o += attn_s[k] * A32[k * DMD + tid];
        o_s[tid] = o;
    }
    __syncthreads();

    // ---------------- Phase 5: W_fc projection + residual + LayerNorm -------
    float accv = 0.0f;
    if (tid < DMD) {
        accv = bfc[tid] + ht_s[tid];
        for (int d = 0; d < DMD; ++d) accv += o_s[d] * Wfc[d * DMD + tid];
    }
    red[tid] = (tid < DMD) ? accv : 0.0f;
    __syncthreads();
    for (int s = 128; s > 0; s >>= 1) {
        if (tid < s) red[tid] += red[tid + s];
        __syncthreads();
    }
    if (tid == 0) stat[0] = red[0] * (1.0f / DMD);
    __syncthreads();
    const float mu = stat[0];
    float dv = (tid < DMD) ? (accv - mu) : 0.0f;
    red[tid] = dv * dv;
    __syncthreads();
    for (int s = 128; s > 0; s >>= 1) {
        if (tid < s) red[tid] += red[tid + s];
        __syncthreads();
    }
    if (tid == 0) stat[1] = red[0] * (1.0f / DMD);
    __syncthreads();
    if (tid < DMD) {
        float rs = rsqrtf(stat[1] + 1e-6f);
        out[rowOff + tid] = (accv - mu) * rs * lns[tid] + lnb[tid];
    }
}

extern "C" void kernel_launch(void* const* d_in, const int* in_sizes, int n_in,
                              void* d_out, int out_size, void* d_ws, size_t ws_size,
                              hipStream_t stream) {
    (void)in_sizes; (void)n_in; (void)out_size; (void)d_ws; (void)ws_size;
    const float* ht     = (const float*)d_in[0];
    const float* factor = (const float*)d_in[1];
    const int*   entn   = (const int*)d_in[2];
    const float* Wcomb  = (const float*)d_in[3];
    const float* bcomb  = (const float*)d_in[4];
    const float* wvv    = (const float*)d_in[5];
    const float* Wfc    = (const float*)d_in[6];
    const float* bfc    = (const float*)d_in[7];
    const float* lns    = (const float*)d_in[8];
    const float* lnb    = (const float*)d_in[9];

    dim3 grid(BSZ * NE * NE);   // 6400 workgroups, one per (b,i,j)
    dim3 block(256);            // 8 wave32 waves
    hipLaunchKernelGGL(sib_attn_fused, grid, block, 0, stream,
                       ht, factor, entn, Wcomb, bcomb, wvv, Wfc, bfc, lns, lnb,
                       (float*)d_out);
}
